// SSTClassifier_29609504539202
// MI455X (gfx1250) — matrix operand order repaired and checked by reading, not compiled, not run on hardware
//
#include <hip/hip_runtime.h>
#include <math.h>

typedef __attribute__((ext_vector_type(16))) __bf16 v16bf;
typedef __attribute__((ext_vector_type(4)))  __bf16 v4bf;
typedef __attribute__((ext_vector_type(8)))  float  v8f;

#define NB   32
#define NT   128
#define NDEP 13
#define DEMB 1024
#define NPART 64
#define NCLS 5
#define DCAP 16
#define NI   (NT*NDEP)      /* 1664 */
#define NTILE (NI/16)       /* 104  */
#define NSLICE 8
#define TPS  (NTILE/NSLICE) /* 13 tiles per slice */

__device__ __forceinline__ float logsig(float a) {
    return (a >= 0.f) ? -log1pf(expf(-a)) : (a - log1pf(expf(a)));
}

// ---------------------------------------------------------------------------
// Pre-swizzle dp_w and W1 into bf16 WMMA B-fragment order:
//   record[fragId][lane][c] , c=0..15 ;  element c of lane:  N = lane&15,
//   K = k0 + (c&7) + 16*(c>>3) + 8*(lane>>4)   (16-bit B 32x16 layout)
// dpf: 4 ntiles x 32 ksteps  -> frag id (nt*32+ks)   (for k_parts)
// w1f: 64 j     x 2  k0sel   -> frag id (j*2+k0sel)  (for k_sweep)
// ---------------------------------------------------------------------------
__global__ __launch_bounds__(256) void k_prep(
    const float* __restrict__ dp_w, const float* __restrict__ W1,
    __bf16* __restrict__ dpf, __bf16* __restrict__ w1f)
{
    const int g = blockIdx.x * 256 + threadIdx.x;   // 8192 threads
    if (g < 4096) {                                  // dp_w fragments
        const int nt = g >> 10, ks = (g >> 5) & 31, lane = g & 31;
        const int n = nt * 16 + (lane & 15);
        v16bf v;
#pragma unroll
        for (int c = 0; c < 16; ++c) {
            int kb = ks * 32 + (c & 7) + ((c >> 3) << 4) + ((lane >> 4) << 3);
            v[c] = (__bf16)dp_w[(size_t)n * DEMB + kb];
        }
        *(v16bf*)(dpf + (size_t)g * 16) = v;
    } else {                                         // W1 fragments
        const int w = g - 4096;
        const int j = w >> 6, k0sel = (w >> 5) & 1, lane = w & 31;
        const int h = lane & 15;
        v16bf v;
#pragma unroll
        for (int c = 0; c < 16; ++c) {
            int d = k0sel * 32 + (c & 7) + ((c >> 3) << 4) + ((lane >> 4) << 3);
            v[c] = (__bf16)W1[((j * NPART) + d) * DCAP + h];
        }
        *(v16bf*)(w1f + (size_t)w * 16) = v;
    }
}

// ---------------------------------------------------------------------------
// Stage A: x = embs + depth_emb; LayerNorm; h = xn @ dp_w^T + dp_b; mu = swish
// xn is written to LDS directly in A-fragment order; output mu goes to global
// in A-fragment order (muA) so k_sweep can vector-load its A operands.
// A 16x32 layout: lane holds row m=lane&15; element c -> K = (c&7)+16*(c>>3)+8*(lane>>4)
// ---------------------------------------------------------------------------
__global__ __launch_bounds__(256) void k_parts(
    const float* __restrict__ embs, const float* __restrict__ depth_emb,
    const float* __restrict__ ln_w, const float* __restrict__ ln_b,
    const __bf16* __restrict__ dpf, const float* __restrict__ dp_b,
    __bf16* __restrict__ muA)
{
    __shared__ __bf16 xnf[32][32][16];   // [kstep][lane][c]  (A-fragment order)
    __shared__ float lw[DEMB], lb[DEMB];

    const int tid  = threadIdx.x;
    const int b    = blockIdx.x / NTILE;
    const int tile = blockIdx.x % NTILE;

    for (int e = tid; e < DEMB; e += 256) { lw[e] = ln_w[e]; lb[e] = ln_b[e]; }

    const int r  = tid >> 4;         // row within tile (0..15)
    const int lc = tid & 15;         // 16 threads per row
    const int i  = tile * 16 + r;
    const int t  = i / NDEP, dd = i % NDEP;
    const float* xrow = embs + (((size_t)b * NT + t) * NDEP + dd) * DEMB;
    const float* drow = depth_emb + (size_t)dd * DEMB;

    float4 xv[16];
    float s = 0.f, ss = 0.f;
#pragma unroll
    for (int q = 0; q < 16; ++q) {
        int col = (lc + 16 * q) * 4;
        float4 e  = *(const float4*)(xrow + col);
        float4 de = *(const float4*)(drow + col);
        float4 x  = make_float4(e.x + de.x, e.y + de.y, e.z + de.z, e.w + de.w);
        xv[q] = x;
        s  += x.x + x.y + x.z + x.w;
        ss += x.x * x.x + x.y * x.y + x.z * x.z + x.w * x.w;
    }
    s  += __shfl_xor(s, 1, 32);  s  += __shfl_xor(s, 2, 32);
    s  += __shfl_xor(s, 4, 32);  s  += __shfl_xor(s, 8, 32);
    ss += __shfl_xor(ss, 1, 32); ss += __shfl_xor(ss, 2, 32);
    ss += __shfl_xor(ss, 4, 32); ss += __shfl_xor(ss, 8, 32);
    const float mean = s * (1.f / DEMB);
    const float var  = ss * (1.f / DEMB) - mean * mean;
    const float rs   = rsqrtf(var + 1e-5f);

    __syncthreads();  // lw/lb visible
#pragma unroll
    for (int q = 0; q < 16; ++q) {
        int col = (lc + 16 * q) * 4;          // 4 consecutive K's, row m = r
        float4 x = xv[q];
        v4bf p;
        p[0] = (__bf16)((x.x - mean) * rs * lw[col + 0] + lb[col + 0]);
        p[1] = (__bf16)((x.y - mean) * rs * lw[col + 1] + lb[col + 1]);
        p[2] = (__bf16)((x.z - mean) * rs * lw[col + 2] + lb[col + 2]);
        p[3] = (__bf16)((x.w - mean) * rs * lw[col + 3] + lb[col + 3]);
        int ks     = col >> 5;
        int lane_a = r + 16 * ((col >> 3) & 1);
        int cb     = (col & 7) + 8 * ((col >> 4) & 1);
        *(v4bf*)(&xnf[ks][lane_a][cb]) = p;
    }
    __syncthreads();

    // WMMA: each wave owns 16 output parts (N tile = wave)
    const int wave = tid >> 5, lane = tid & 31;
    const int mm = lane & 15, hi = lane >> 4;
    const int n  = wave * 16 + mm;
    v8f acc = {};
    for (int ks = 0; ks < 32; ++ks) {
        v16bf af  = *(const v16bf*)(&xnf[ks][lane][0]);
        v16bf bfr = *(const v16bf*)(dpf + (size_t)((wave * 32 + ks) * 32 + lane) * 16);
        acc = __builtin_amdgcn_wmma_f32_16x16x32_bf16(false, af, false, bfr,
                                                      (short)0, acc, false, false);
    }
    const float bia = dp_b[n];
    // store mu in A-fragment order: k = n (input dim of routing-1 matmul)
    const int kk     = n & 31;
    const int k0sel  = n >> 5;
    const int hi_a   = (kk >> 3) & 1;
    const int c_a    = (kk & 7) + 8 * ((kk >> 4) & 1);
#pragma unroll
    for (int rr = 0; rr < 8; ++rr) {
        float h   = acc[rr] + bia;
        float muv = h / (1.f + expf(-h));            // swish
        int m      = rr + (hi << 3);                 // D layout: M = r + 8*half
        int lane_a = m + 16 * hi_a;
        muA[(size_t)(((b * NTILE + tile) * 2 + k0sel) * 32 + lane_a) * 16 + c_a]
            = (__bf16)muv;
    }
}

// ---------------------------------------------------------------------------
// s_fa[b] = sum_i sigmoid(logit(mask)) = 13 * sum_t mask[b,t]
// ---------------------------------------------------------------------------
__global__ void k_fa(const float* __restrict__ mask, float* __restrict__ s_fa)
{
    __shared__ float red[128];
    const int b = blockIdx.x, tid = threadIdx.x;
    red[tid] = mask[b * NT + tid];
    __syncthreads();
    for (int o = 64; o > 0; o >>= 1) {
        if (tid < o) red[tid] += red[tid + o];
        __syncthreads();
    }
    if (tid == 0) s_fa[b] = (float)NDEP * red[0];
}

// ---------------------------------------------------------------------------
// Routing-1 sweep: recompute V tile by WMMA (all operands pre-swizzled, loaded
// as 32B vectors), E-step softmax, accumulate weighted moments.
// One block per (b, i-slice); deterministic per-slice partials.
// ---------------------------------------------------------------------------
__global__ __launch_bounds__(256) void k_sweep(
    const __bf16* __restrict__ muA, const __bf16* __restrict__ w1f,
    const float* __restrict__ mask, const float* __restrict__ B1,
    const float* __restrict__ mu_g, const float* __restrict__ i2s_g,
    const float* __restrict__ c_g,
    float* __restrict__ gD, float* __restrict__ gS, float* __restrict__ gQ,
    int iter)
{
    __shared__ __bf16 Vt[16][NPART * DCAP];   // 16 x 1024 V tile (bf16)
    __shared__ float lp[16][NPART];
    __shared__ float Du[16][NPART];
    __shared__ float aD[NPART];
    __shared__ float aS[NPART][DCAP];
    __shared__ float aQ[NPART][DCAP];
    __shared__ float mu_s[NPART][DCAP];
    __shared__ float i2s[NPART][DCAP];
    __shared__ float cjl[NPART];

    const int b = blockIdx.x, slice = blockIdx.y;
    const int tid = threadIdx.x, wave = tid >> 5, lane = tid & 31;
    const int m = lane & 15, hi = lane >> 4;

    for (int e = tid; e < NPART * DCAP; e += 256) {
        aS[e >> 4][e & 15] = 0.f; aQ[e >> 4][e & 15] = 0.f;
    }
    if (tid < NPART) {
        aD[tid] = 0.f;
        if (iter > 0) cjl[tid] = c_g[b * NPART + tid];
    }
    if (iter > 0) {
        for (int e = tid; e < NPART * DCAP; e += 256) {
            mu_s[e >> 4][e & 15] = mu_g[b * NPART * DCAP + e];
            i2s [e >> 4][e & 15] = i2s_g[b * NPART * DCAP + e];
        }
    }
    __syncthreads();

    for (int it = 0; it < TPS; ++it) {
        const int itile = slice * TPS + it;
        // A fragments: 2 vector loads from L2-hot fragment-ordered muA
        const __bf16* aBase =
            muA + (size_t)((b * NTILE + itile) * 2) * 32 * 16 + (size_t)lane * 16;
        v16bf a0 = *(const v16bf*)(aBase);
        v16bf a1 = *(const v16bf*)(aBase + 32 * 16);

        const int h = m;  // this lane's d_out column
        for (int jj = 0; jj < 16; ++jj) {
            const int j = wave * 16 + jj;
            const __bf16* bBase = w1f + (size_t)(j * 2 * 32 + lane) * 16;
            v16bf b0 = *(const v16bf*)(bBase);
            v16bf b1 = *(const v16bf*)(bBase + 32 * 16);
            v8f acc = {};
            acc = __builtin_amdgcn_wmma_f32_16x16x32_bf16(false, a0, false, b0,
                                                          (short)0, acc, false, false);
            acc = __builtin_amdgcn_wmma_f32_16x16x32_bf16(false, a1, false, b1,
                                                          (short)0, acc, false, false);
            const float bj = B1[j * DCAP + h];
            float vr[8];
#pragma unroll
            for (int rr = 0; rr < 8; ++rr) {
                float v = acc[rr] + bj;
                vr[rr] = v;
                Vt[rr + (hi << 3)][j * DCAP + h] = (__bf16)v;
            }
            if (iter > 0) {
#pragma unroll
                for (int rr = 0; rr < 8; ++rr) {
                    float dv = vr[rr] - mu_s[j][h];
                    float p  = dv * dv * i2s[j][h];
                    p += __shfl_xor(p, 1, 32); p += __shfl_xor(p, 2, 32);
                    p += __shfl_xor(p, 4, 32); p += __shfl_xor(p, 8, 32);
                    if (m == 0) lp[rr + (hi << 3)][j] = -p;
                }
            }
        }
        __syncthreads();
        // ---- E-step: softmax over j, D_use = fa * R ----
        {
            const int i  = tid >> 4;
            const int jb = (tid & 15) * 4;
            const int ig = itile * 16 + i;
            const float fa = mask[b * NT + ig / NDEP];  // sigmoid(logit(m)) == m
            if (iter == 0) {
#pragma unroll
                for (int jj = 0; jj < 4; ++jj) Du[i][jb + jj] = fa * (1.f / NPART);
            } else {
                float f[4], mx = -3.4e38f;
#pragma unroll
                for (int jj = 0; jj < 4; ++jj) {
                    f[jj] = lp[i][jb + jj] + cjl[jb + jj];
                    mx = fmaxf(mx, f[jj]);
                }
                mx = fmaxf(mx, __shfl_xor(mx, 1, 32));
                mx = fmaxf(mx, __shfl_xor(mx, 2, 32));
                mx = fmaxf(mx, __shfl_xor(mx, 4, 32));
                mx = fmaxf(mx, __shfl_xor(mx, 8, 32));
                float e[4], sum = 0.f;
#pragma unroll
                for (int jj = 0; jj < 4; ++jj) { e[jj] = expf(f[jj] - mx); sum += e[jj]; }
                sum += __shfl_xor(sum, 1, 32); sum += __shfl_xor(sum, 2, 32);
                sum += __shfl_xor(sum, 4, 32); sum += __shfl_xor(sum, 8, 32);
                float inv = fa / sum;
#pragma unroll
                for (int jj = 0; jj < 4; ++jj) Du[i][jb + jj] = e[jj] * inv;
            }
        }
        __syncthreads();
        // ---- accumulate weighted moments (each (j,h) owned by one thread) ----
        {
            const int j  = tid >> 2;
            const int hb = (tid & 3) * 4;
            float sm[4] = {0, 0, 0, 0}, sq[4] = {0, 0, 0, 0}, sd = 0.f;
            for (int i = 0; i < 16; ++i) {
                float d = Du[i][j];
                sd += d;
#pragma unroll
                for (int hh = 0; hh < 4; ++hh) {
                    float v = (float)Vt[i][j * DCAP + hb + hh];
                    sm[hh] += d * v;
                    sq[hh] += d * v * v;
                }
            }
#pragma unroll
            for (int hh = 0; hh < 4; ++hh) { aS[j][hb + hh] += sm[hh]; aQ[j][hb + hh] += sq[hh]; }
            if ((tid & 3) == 0) aD[j] += sd;
        }
        __syncthreads();
    }
    // deterministic per-slice partials (no atomics)
    {
        const int j  = tid >> 2;
        const int hb = (tid & 3) * 4;
        const size_t base = ((size_t)(b * NSLICE + slice) * NPART + j);
#pragma unroll
        for (int hh = 0; hh < 4; ++hh) {
            gS[base * DCAP + hb + hh] = aS[j][hb + hh];
            gQ[base * DCAP + hb + hh] = aQ[j][hb + hh];
        }
        if ((tid & 3) == 0) gD[base] = aD[j];
    }
}

// ---------------------------------------------------------------------------
// Routing-1 M-step finalize: sum slice partials, a_out, mu, 1/(2sig2), c_j
// ---------------------------------------------------------------------------
__global__ void k_fin1(
    const float* __restrict__ gD, const float* __restrict__ gS,
    const float* __restrict__ gQ, const float* __restrict__ s_fa,
    const float* __restrict__ bu, const float* __restrict__ bi,
    float* __restrict__ a1, float* __restrict__ mu1,
    float* __restrict__ i2s, float* __restrict__ c1)
{
    const int b = blockIdx.x, j = threadIdx.x;  // 64 threads
    float D = 0.f;
    for (int s = 0; s < NSLICE; ++s) D += gD[(size_t)(b * NSLICE + s) * NPART + j];
    const float sfa = s_fa[b];
    const float a   = bu[j] * D - bi[j] * (sfa - D);
    const float invD = 1.f / D;
    float logdet = 0.f;
    for (int h = 0; h < DCAP; ++h) {
        float S = 0.f, Q = 0.f;
        for (int s = 0; s < NSLICE; ++s) {
            size_t idx = ((size_t)(b * NSLICE + s) * NPART + j) * DCAP + h;
            S += gS[idx]; Q += gQ[idx];
        }
        float mu = S * invD;
        float s2 = Q * invD - mu * mu + 1e-5f;
        mu1[(b * NPART + j) * DCAP + h] = mu;
        i2s[(b * NPART + j) * DCAP + h] = 0.5f / s2;
        logdet += logf(s2);
    }
    a1[b * NPART + j] = a;
    c1[b * NPART + j] = logsig(a) - 0.5f * logdet;
}

// ---------------------------------------------------------------------------
// Routing 2 (tiny): one block per batch, everything in LDS.
// ---------------------------------------------------------------------------
__global__ __launch_bounds__(128) void k_route2(
    const float* __restrict__ a1, const float* __restrict__ mu1,
    const float* __restrict__ W2, const float* __restrict__ B2,
    const float* __restrict__ bu2, const float* __restrict__ bi2,
    float* __restrict__ out)
{
    __shared__ float V2[NPART][NCLS * DCAP];
    __shared__ float fa2[NPART];
    __shared__ float Du[NPART][NCLS];
    __shared__ float mu2[NCLS][DCAP], i2[NCLS][DCAP], s2l[NCLS][DCAP];
    __shared__ float Sm[NCLS][DCAP], Sq[NCLS][DCAP];
    __shared__ float cj[NCLS], a2[NCLS], Dsm[NCLS];
    __shared__ float sfa2s;

    const int b = blockIdx.x, tid = threadIdx.x;
    for (int e = tid; e < NPART * NCLS * DCAP; e += 128) {
        int i = e / (NCLS * DCAP), rem = e % (NCLS * DCAP);
        int j = rem / DCAP, h = rem % DCAP;
        float acc = B2[j * DCAP + h];
        const float* mrow = mu1 + (size_t)(b * NPART + i) * DCAP;
        const float* wrow = W2 + ((size_t)(i * NCLS + j) * DCAP) * DCAP + h;
        for (int d = 0; d < DCAP; ++d) acc += wrow[d * DCAP] * mrow[d];
        V2[i][rem] = acc;
    }
    if (tid < NPART) fa2[tid] = 1.f / (1.f + expf(-a1[b * NPART + tid]));
    __syncthreads();
    if (tid == 0) { float s = 0.f; for (int i = 0; i < NPART; ++i) s += fa2[i]; sfa2s = s; }
    __syncthreads();

    for (int iter = 0; iter < 3; ++iter) {
        if (tid < NPART) {
            const int i = tid;
            if (iter == 0) {
                for (int j = 0; j < NCLS; ++j) Du[i][j] = fa2[i] * (1.f / NCLS);
            } else {
                float f[NCLS], mx = -3.4e38f;
                for (int j = 0; j < NCLS; ++j) {
                    float p = 0.f;
                    for (int h = 0; h < DCAP; ++h) {
                        float dv = V2[i][j * DCAP + h] - mu2[j][h];
                        p += dv * dv * i2[j][h];
                    }
                    f[j] = cj[j] - p;
                    mx = fmaxf(mx, f[j]);
                }
                float sum = 0.f, e[NCLS];
                for (int j = 0; j < NCLS; ++j) { e[j] = expf(f[j] - mx); sum += e[j]; }
                float inv = fa2[i] / sum;
                for (int j = 0; j < NCLS; ++j) Du[i][j] = e[j] * inv;
            }
        }
        __syncthreads();
        if (tid < NCLS * DCAP) {
            int j = tid / DCAP, h = tid % DCAP;
            float sm = 0.f, sq = 0.f;
            for (int i = 0; i < NPART; ++i) {
                float d = Du[i][j], v = V2[i][j * DCAP + h];
                sm += d * v; sq += d * v * v;
            }
            Sm[j][h] = sm; Sq[j][h] = sq;
        } else if (tid < NCLS * DCAP + NCLS) {
            int j = tid - NCLS * DCAP;
            float s = 0.f;
            for (int i = 0; i < NPART; ++i) s += Du[i][j];
            Dsm[j] = s;
        }
        __syncthreads();
        if (tid < NCLS) {
            const int j = tid;
            float D = Dsm[j];
            float a = bu2[j] * D - bi2[j] * (sfa2s - D);
            float invD = 1.f / D, logdet = 0.f;
            for (int h = 0; h < DCAP; ++h) {
                float mu = Sm[j][h] * invD;
                float s2 = Sq[j][h] * invD - mu * mu + 1e-5f;
                mu2[j][h] = mu; s2l[j][h] = s2; i2[j][h] = 0.5f / s2;
                logdet += logf(s2);
            }
            a2[j] = a;
            cj[j] = logsig(a) - 0.5f * logdet;
        }
        __syncthreads();
    }
    if (tid < NCLS) out[b * NCLS + tid] = a2[tid];
    if (tid < NCLS * DCAP) {
        int j = tid / DCAP, h = tid % DCAP;
        out[NB * NCLS + (b * NCLS + j) * DCAP + h] = mu2[j][h];
        out[NB * NCLS + NB * NCLS * DCAP + (b * NCLS + j) * DCAP + h] = s2l[j][h];
    }
}

// ---------------------------------------------------------------------------
extern "C" void kernel_launch(void* const* d_in, const int* in_sizes, int n_in,
                              void* d_out, int out_size, void* d_ws, size_t ws_size,
                              hipStream_t stream)
{
    (void)in_sizes; (void)n_in; (void)out_size; (void)ws_size;
    const float* mask      = (const float*)d_in[0];
    const float* embs      = (const float*)d_in[1];
    const float* depth_emb = (const float*)d_in[2];
    const float* ln_w      = (const float*)d_in[3];
    const float* ln_b      = (const float*)d_in[4];
    const float* dp_w      = (const float*)d_in[5];
    const float* dp_b      = (const float*)d_in[6];
    const float* W1        = (const float*)d_in[7];
    const float* B1        = (const float*)d_in[8];
    const float* bu1       = (const float*)d_in[9];
    const float* bi1       = (const float*)d_in[10];
    const float* W2        = (const float*)d_in[11];
    const float* B2        = (const float*)d_in[12];
    const float* bu2       = (const float*)d_in[13];
    const float* bi2       = (const float*)d_in[14];
    float* out = (float*)d_out;

    char* ws = (char*)d_ws;
    size_t off = 0;
    auto carve = [&](size_t bytes) -> void* {
        void* p = ws + off;
        off = (off + bytes + 255) & ~(size_t)255;
        return p;
    };
    float*  s_fa  = (float*)carve(NB * sizeof(float));
    float*  gD    = (float*)carve((size_t)NB * NSLICE * NPART * sizeof(float));
    float*  gS    = (float*)carve((size_t)NB * NSLICE * NPART * DCAP * sizeof(float));
    float*  gQ    = (float*)carve((size_t)NB * NSLICE * NPART * DCAP * sizeof(float));
    float*  a1    = (float*)carve((size_t)NB * NPART * sizeof(float));
    float*  mu1   = (float*)carve((size_t)NB * NPART * DCAP * sizeof(float));
    float*  i2s   = (float*)carve((size_t)NB * NPART * DCAP * sizeof(float));
    float*  c1    = (float*)carve((size_t)NB * NPART * sizeof(float));
    __bf16* dpf   = (__bf16*)carve((size_t)4 * 32 * 32 * 16 * sizeof(__bf16));
    __bf16* w1f   = (__bf16*)carve((size_t)NPART * 2 * 32 * 16 * sizeof(__bf16));
    __bf16* muA   = (__bf16*)carve((size_t)NB * NTILE * 2 * 32 * 16 * sizeof(__bf16));

    k_prep<<<32, 256, 0, stream>>>(dp_w, W1, dpf, w1f);
    k_parts<<<NB * NTILE, 256, 0, stream>>>(embs, depth_emb, ln_w, ln_b,
                                            dpf, dp_b, muA);
    k_fa<<<NB, 128, 0, stream>>>(mask, s_fa);
    for (int iter = 0; iter < 3; ++iter) {
        k_sweep<<<dim3(NB, NSLICE), 256, 0, stream>>>(
            muA, w1f, mask, B1, mu1, i2s, c1, gD, gS, gQ, iter);
        k_fin1<<<NB, NPART, 0, stream>>>(gD, gS, gQ, s_fa, bu1, bi1,
                                         a1, mu1, i2s, c1);
    }
    k_route2<<<NB, 128, 0, stream>>>(a1, mu1, W2, B2, bu2, bi2, out);
}
